// afVSRN_89799176225529
// MI455X (gfx1250) — compile-verified, matrix operand order checked by reading, not run on hardware
//
#include <hip/hip_runtime.h>
#include <hip/hip_bf16.h>
#include <math.h>

typedef __attribute__((ext_vector_type(16))) _Float16 v16h;
typedef __attribute__((ext_vector_type(8)))  float    v8f;

#define N_PTS    131072
#define N_FEATS  1000
#define FPAD     1024
#define FDIM     64
#define HID      128
#define PE_N     36
#define NCHUNK   32           // FPAD/32 K-chunks for feature stage
#define NTHREADS 512          // 16 waves -> 4 waves per SIMD32
#define WAVES    16
#define PTS_PER_BLOCK (WAVES * 16)          // 256
#define NBLOCKS  (N_PTS / PTS_PER_BLOCK)    // 512

// ---- LDS layout (bytes) ----
#define XS_OFF   0                       // 256*3 f32                 = 3072
#define LOC4_OFF 3072                    // 1000 float4 (2l, ||l||^2) = 16384 (pad)
#define INVS_OFF (LOC4_OFF + 16384)      // 256 f32                   = 1024
#define BIAS_OFF (INVS_OFF + 1024)       // b0..b3 + W4 = 5*128 f32   = 2560
#define BIG_OFF  (BIAS_OFF + 2560)       // 128 KB: fv frags, later W0..W3 frags
#define ACT_OFF  (BIG_OFF + 131072)      // 2 bufs * 16 waves * 4 KB  = 128 KB
#define SMEM_BYTES (ACT_OFF + 131072)    // 285184 B  (< 320 KB WGP LDS)

#define LOG2E 1.44269504088896f
#define PI_F  3.14159265358979f

// ISA 16-bit A/B fragment K mapping (CDNA5 ISA §7.12.2):
// lanes0-15 : regs0-3 -> K=0..7,  regs4-7 -> K=16..23
// lanes16-31: regs0-3 -> K=8..15, regs4-7 -> K=24..31
__device__ __forceinline__ int kmap(int e, int h) {
    return (e & 7) + ((e >> 3) << 4) + (h << 3);
}
// inverse: slot (half, elem) that holds K within a 32-wide chunk
__device__ __forceinline__ void kinv(int k, int& half, int& e) {
    half = (k >> 3) & 1;
    e    = (k & 7) + ((k & 16) ? 8 : 0);
}

__device__ __forceinline__ v8f wmma_f16(v16h a, v16h b, v8f c) {
    return __builtin_amdgcn_wmma_f32_16x16x32_f16(false, a, false, b,
                                                  (short)0, c, false, false);
}

// fast d^2 -> unnormalized softmax weight (v_sqrt + v_rcp + v_exp)
__device__ __forceinline__ float wfast(float d2, float negmxl) {
    float inv = __builtin_amdgcn_rcpf(__builtin_amdgcn_sqrtf(d2) + 1e-6f);
    return __builtin_amdgcn_exp2f(fmaf(inv, LOG2E, negmxl));
}

// one 128->128 layer: 8 N-tiles x 4 K-chunks of WMMA, bias + SnakeAlt, scatter
// result back into A-fragment order for the next layer.
__device__ __forceinline__ void mlp_layer(const _Float16* Ain, _Float16* Aout,
                                          const v16h* WB, const float* biasLds,
                                          int lane, int mr, int hh) {
    const v16h* AV = (const v16h*)Ain;
    // hoist A fragments: reused by all 8 output tiles
    v16h a0 = AV[0 * 32 + lane], a1 = AV[1 * 32 + lane];
    v16h a2 = AV[2 * 32 + lane], a3 = AV[3 * 32 + lane];
    for (int t = 0; t < 8; ++t) {
        v16h w0 = WB[(0 * 8 + t) * 32 + lane];
        v16h w1 = WB[(1 * 8 + t) * 32 + lane];
        v16h w2 = WB[(2 * 8 + t) * 32 + lane];
        v16h w3 = WB[(3 * 8 + t) * 32 + lane];
        v8f cc = {};
        cc = wmma_f16(a0, w0, cc);
        cc = wmma_f16(a1, w1, cc);
        cc = wmma_f16(a2, w2, cc);
        cc = wmma_f16(a3, w3, cc);
        const int N = t * 16 + mr;            // output column this lane holds
        const float b = biasLds[N];
        int hp, e2; kinv(N & 31, hp, e2);
        const int c2 = N >> 5;
#pragma unroll
        for (int r = 0; r < 8; ++r) {         // C layout: row M = r + 8*half
            float h = cc[r] + b;
            float s = __sinf(h);
            float v = 0.5f * h + s * s;       // SnakeAlt
            Aout[(c2 * 32 + (r + 8 * hh) + 16 * hp) * 16 + e2] = (_Float16)v;
        }
    }
}

__global__ __launch_bounds__(NTHREADS, 1)
void afvsrn_fused(const float* __restrict__ x,
                  const float* __restrict__ floc,
                  const float* __restrict__ fvec,
                  const float* __restrict__ W0, const float* __restrict__ b0,
                  const float* __restrict__ W1, const float* __restrict__ b1,
                  const float* __restrict__ W2, const float* __restrict__ b2,
                  const float* __restrict__ W3, const float* __restrict__ b3,
                  const float* __restrict__ W4, const float* __restrict__ b4,
                  float* __restrict__ out) {
    extern __shared__ char smem[];
    float*    xs   = (float*)(smem + XS_OFF);
    float4*   loc4 = (float4*)(smem + LOC4_OFF);
    float*    invs = (float*)(smem + INVS_OFF);
    float*    bseg = (float*)(smem + BIAS_OFF);   // [0..511]=b0..b3, [512..639]=W4
    _Float16* big  = (_Float16*)(smem + BIG_OFF);
    _Float16* act  = (_Float16*)(smem + ACT_OFF);

    const int tid  = threadIdx.x;
    const int lane = tid & 31;
    const int wv   = tid >> 5;
    const int hh   = lane >> 4;      // which half of the wave
    const int mr   = lane & 15;      // row within 16-pt tile / col within N-tile
    const int pl   = wv * 16 + mr;   // point index within block
    const int p    = blockIdx.x * PTS_PER_BLOCK + pl;

    // ---------- stage A: stage inputs into LDS ----------
    for (int i = tid; i < PTS_PER_BLOCK * 3; i += NTHREADS)
        xs[i] = x[blockIdx.x * PTS_PER_BLOCK * 3 + i];
    for (int i = tid; i < N_FEATS; i += NTHREADS) {
        float a0 = floc[i * 3], a1 = floc[i * 3 + 1], a2 = floc[i * 3 + 2];
        float4 v; v.x = 2.0f * a0; v.y = 2.0f * a1; v.z = 2.0f * a2;
        v.w = a0 * a0 + a1 * a1 + a2 * a2;
        loc4[i] = v;
    }
    for (int i = tid; i < 5 * HID; i += NTHREADS) {   // biases + W4 to LDS
        int l = i >> 7, j = i & 127;
        const float* src = (l == 0) ? b0 : (l == 1) ? b1 : (l == 2) ? b2
                                                   : (l == 3) ? b3 : W4;
        bseg[i] = src[j];
    }
    // feature_vectors: COALESCED row-major global reads, scatter to B-fragment
    // order in LDS.  value fvec[f][j] -> slot ((c*4+t)*32 + ln)*16 + e
    for (int s = tid; s < FPAD * FDIM; s += NTHREADS) {
        int f = s >> 6, j = s & 63;
        float v = (f < N_FEATS) ? fvec[s] : 0.0f;     // fvec[f*64 + j], coalesced
        int c = f >> 5, hp, e; kinv(f & 31, hp, e);
        int ln = 16 * hp + (j & 15), t = j >> 4;
        big[((c * 4 + t) * 32 + ln) * 16 + e] = (_Float16)v;
    }
    __syncthreads();

    const float x0 = xs[pl * 3], x1 = xs[pl * 3 + 1], x2 = xs[pl * 3 + 2];
    const float xx = x0 * x0 + x1 * x1 + x2 * x2;

    // ---------- stage B pass 1: min d^2  (max inv_d == inv_d(min d^2)) -------
    float dmin = 1e30f;
    for (int f = hh; f < N_FEATS; f += 2) {
        float4 l = loc4[f];
        float t = xx + l.w;
        t = fmaf(-x0, l.x, t); t = fmaf(-x1, l.y, t); t = fmaf(-x2, l.z, t);
        dmin = fminf(dmin, t);
    }
    dmin = fminf(dmin, __shfl_xor(dmin, 16));
    dmin = fmaxf(dmin, 0.0f);
    const float mx = __builtin_amdgcn_rcpf(__builtin_amdgcn_sqrtf(dmin) + 1e-6f);
    const float negmxl = -mx * LOG2E;

    // ---------- stage B pass 2: WMMA  w_exp @ feature_vectors ----------
    v8f acc0 = {}, acc1 = {}, acc2 = {}, acc3 = {};
    float ssum = 0.0f;
    const v16h* BV = (const v16h*)big;
    for (int c = 0; c < NCHUNK - 1; ++c) {        // chunks 0..30: f <= 991, no guard
        // hoist B fragments so LDS latency hides under the A-fragment VALU work
        v16h bf0 = BV[(c * 4 + 0) * 32 + lane];
        v16h bf1 = BV[(c * 4 + 1) * 32 + lane];
        v16h bf2 = BV[(c * 4 + 2) * 32 + lane];
        v16h bf3 = BV[(c * 4 + 3) * 32 + lane];
        v16h a;
#pragma unroll
        for (int e = 0; e < 16; ++e) {
            float4 l = loc4[c * 32 + kmap(e, hh)];
            float t = xx + l.w;
            t = fmaf(-x0, l.x, t); t = fmaf(-x1, l.y, t); t = fmaf(-x2, l.z, t);
            float w = wfast(fmaxf(t, 0.0f), negmxl);
            ssum += w;
            a[e] = (_Float16)w;
        }
        acc0 = wmma_f16(a, bf0, acc0);
        acc1 = wmma_f16(a, bf1, acc1);
        acc2 = wmma_f16(a, bf2, acc2);
        acc3 = wmma_f16(a, bf3, acc3);
    }
    {   // tail chunk c=31: only f=992..999 live -> hh==0, e<8
        const int c = NCHUNK - 1;
        v16h bf0 = BV[(c * 4 + 0) * 32 + lane];
        v16h bf1 = BV[(c * 4 + 1) * 32 + lane];
        v16h bf2 = BV[(c * 4 + 2) * 32 + lane];
        v16h bf3 = BV[(c * 4 + 3) * 32 + lane];
        v16h a = {};
        if (hh == 0) {
#pragma unroll
            for (int e = 0; e < 8; ++e) {         // kmap(e,0)=e -> f=992+e<1000
                float4 l = loc4[c * 32 + e];
                float t = xx + l.w;
                t = fmaf(-x0, l.x, t); t = fmaf(-x1, l.y, t); t = fmaf(-x2, l.z, t);
                float w = wfast(fmaxf(t, 0.0f), negmxl);
                ssum += w;
                a[e] = (_Float16)w;
            }
        }
        acc0 = wmma_f16(a, bf0, acc0);            // EXEC all-1s again here
        acc1 = wmma_f16(a, bf1, acc1);
        acc2 = wmma_f16(a, bf2, acc2);
        acc3 = wmma_f16(a, bf3, acc3);
    }
    float stot = ssum + __shfl_xor(ssum, 16);
    if (hh == 0) invs[wv * 16 + mr] = __builtin_amdgcn_rcpf(stot);
    __syncthreads();

    // ---------- stage C: scatter feats + PE into A-fragment layer-0 input ----
    _Float16* A0 = act + (0 * WAVES + wv) * 2048;
    {
        v8f accs[4] = {acc0, acc1, acc2, acc3};
#pragma unroll
        for (int t = 0; t < 4; ++t) {
            const int K = PE_N + t * 16 + mr;     // 36..99
            int hp, e2; kinv(K & 31, hp, e2);
            const int c2 = K >> 5;
#pragma unroll
            for (int r = 0; r < 8; ++r) {
                const int M = r + 8 * hh;
                float v = accs[t][r] * invs[wv * 16 + M];
                A0[(c2 * 32 + M + 16 * hp) * 16 + e2] = (_Float16)v;
            }
        }
        // PE for K<36, zero-pad 100..127 (this lane owns its own slots)
#pragma unroll
        for (int c = 0; c < 4; ++c)
#pragma unroll
            for (int e = 0; e < 16; ++e) {
                int K = c * 32 + kmap(e, hh);
                if (K >= PE_N && K < 100) continue;   // filled by feats
                float v = 0.0f;
                if (K < PE_N) {
                    int d = K / 12, r = K % 12, l = r % 6;
                    float xd = (d == 0) ? x0 : (d == 1) ? x1 : x2;
                    float ang = xd * ((float)(1 << l) * PI_F);
                    v = (r < 6) ? __sinf(ang) : __cosf(ang);
                }
                A0[(c * 32 + lane) * 16 + e] = (_Float16)v;
            }
    }
    __syncthreads();

    // ---------- stage C': MLP weights, coalesced reads -> B-fragments --------
    // value W_l[j][k] -> slot l*16384 + ((c*8+t)*32 + ln)*16 + e
    for (int s = tid; s < 4 * HID * HID; s += NTHREADS) {
        int l = s >> 14, r = s & 16383;
        int j = r >> 7, k = r & 127;              // consecutive k -> coalesced
        float v;
        if (l == 0)      v = (k < 100) ? W0[j * 100 + k] : 0.0f;
        else if (l == 1) v = W1[j * HID + k];
        else if (l == 2) v = W2[j * HID + k];
        else             v = W3[j * HID + k];
        int c = k >> 5, hp, e; kinv(k & 31, hp, e);
        int ln = 16 * hp + (j & 15), t = j >> 4;
        big[l * 16384 + ((c * 8 + t) * 32 + ln) * 16 + e] = (_Float16)v;
    }
    __syncthreads();

    // ---------- stage D: 4 hidden layers, all WMMA ----------
    _Float16* Abuf0 = act + (0 * WAVES + wv) * 2048;
    _Float16* Abuf1 = act + (1 * WAVES + wv) * 2048;
    mlp_layer(Abuf0, Abuf1, (const v16h*)(big + 0 * 16384), bseg + 0 * HID, lane, mr, hh);
    __syncthreads();
    mlp_layer(Abuf1, Abuf0, (const v16h*)(big + 1 * 16384), bseg + 1 * HID, lane, mr, hh);
    __syncthreads();
    mlp_layer(Abuf0, Abuf1, (const v16h*)(big + 2 * 16384), bseg + 2 * HID, lane, mr, hh);
    __syncthreads();
    mlp_layer(Abuf1, Abuf0, (const v16h*)(big + 3 * 16384), bseg + 3 * HID, lane, mr, hh);
    __syncthreads();

    // ---------- final 128 -> 1 layer (VALU dot, wave32 cross-half reduce) ----
    const float* w4s = bseg + 4 * HID;
    const v16h* AF = (const v16h*)Abuf0;
    float o = 0.0f;
#pragma unroll
    for (int c = 0; c < 4; ++c) {
        v16h av = AF[c * 32 + lane];
#pragma unroll
        for (int e = 0; e < 16; ++e) {
            int K = c * 32 + kmap(e, hh);
            o = fmaf((float)av[e], w4s[K], o);
        }
    }
    o += __shfl_xor(o, 16);
    if (hh == 0) out[p] = o + b4[0];
}

extern "C" void kernel_launch(void* const* d_in, const int* in_sizes, int n_in,
                              void* d_out, int out_size, void* d_ws, size_t ws_size,
                              hipStream_t stream) {
    (void)in_sizes; (void)n_in; (void)out_size; (void)d_ws; (void)ws_size;
    const float* x    = (const float*)d_in[0];
    const float* floc = (const float*)d_in[1];
    const float* fvec = (const float*)d_in[2];
    const float* W0 = (const float*)d_in[3];  const float* b0 = (const float*)d_in[4];
    const float* W1 = (const float*)d_in[5];  const float* b1 = (const float*)d_in[6];
    const float* W2 = (const float*)d_in[7];  const float* b2 = (const float*)d_in[8];
    const float* W3 = (const float*)d_in[9];  const float* b3 = (const float*)d_in[10];
    const float* W4 = (const float*)d_in[11]; const float* b4 = (const float*)d_in[12];
    float* out = (float*)d_out;

    // opt into >64KB dynamic LDS (not a stream op; graph-capture safe)
    static int once = [] {
        hipFuncSetAttribute((const void*)afvsrn_fused,
                            hipFuncAttributeMaxDynamicSharedMemorySize, SMEM_BYTES);
        return 0;
    }();
    (void)once;

    afvsrn_fused<<<dim3(NBLOCKS), dim3(NTHREADS), SMEM_BYTES, stream>>>(
        x, floc, fvec, W0, b0, W1, b1, W2, b2, W3, b3, W4, b4, out);
}
